// ModifiedMultiheadAttention_31482110279706
// MI455X (gfx1250) — compile-verified, matrix operand order checked by reading, not run on hardware
//
#include <hip/hip_runtime.h>
#include <hip/hip_bf16.h>

typedef __attribute__((ext_vector_type(16))) _Float16 v16h;
typedef __attribute__((ext_vector_type(8)))  float    v8f;

constexpr int S  = 2048;
constexpr int Bz = 4;
constexpr int E  = 512;
constexpr int H  = 8;
constexpr int HD = 64;
constexpr int NBH = Bz * H;   // 32 head-batches
constexpr int M   = S * Bz;   // 8192 rows in the projection GEMMs

__device__ __forceinline__ v8f wmma_f16(v16h a, v16h b, v8f c) {
  // D = A(16x32 f16) * B(32x16 f16) + C(16x16 f32)
  return __builtin_amdgcn_wmma_f32_16x16x32_f16(
      /*neg_a=*/false, a, /*neg_b=*/false, b,
      /*c_mod=*/(short)0, c, /*reuse_a=*/false, /*reuse_b=*/false);
}

// Load 16 consecutive f32 and convert to a packed f16 fragment chunk.
__device__ __forceinline__ v16h cvt_row16(const float* __restrict__ p) {
  v16h r;
#pragma unroll
  for (int i = 0; i < 16; ++i) r[i] = (_Float16)p[i];
  return r;
}

// ---------------------------------------------------------------------------
// Kernel 1: fused QKV input projections; one wave owns a 32x32 output tile
// (4 independent accumulators -> 4 back-to-back independent WMMAs per K-step,
//  halves fragment-load traffic per WMMA vs a 16x16 tile).
//   z=0: Q = (query @ w_qk^T + b_qk) * 0.125   -> Qw[bh][s][d]   (f16)
//   z=1: K =  key   @ w_qk^T + b_qk            -> Kw[bh][s][d]   (f16)
//   z=2: V =  value @ w_v^T  + b_v             -> Vt[bh][d][s]   (f16, transposed)
// ---------------------------------------------------------------------------
__global__ __launch_bounds__(128)
void qkv_proj_kernel(const float* __restrict__ q_in, const float* __restrict__ k_in,
                     const float* __restrict__ v_in,
                     const float* __restrict__ ipw, const float* __restrict__ ipb,
                     _Float16* __restrict__ Qw, _Float16* __restrict__ Kw,
                     _Float16* __restrict__ Vt) {
  const int lane  = threadIdx.x & 31;
  const int wave  = threadIdx.x >> 5;
  const int mt    = blockIdx.x;               // 0..255  (32-row tile)
  const int nt    = blockIdx.y * 4 + wave;    // 0..15   (32-col tile)
  const int z     = blockIdx.z;               // 0..2
  const int lrow  = lane & 15;
  const int khalf = (lane >> 4) * 16;

  const float* X    = (z == 0) ? q_in : (z == 1) ? k_in : v_in;
  const float* W    = ipw + (z == 2 ? (size_t)2 * E * E : 0);
  const float* bias = ipb + (z == 2 ? 2 * E : 0);

  const float* xrow0 = X + (size_t)(mt * 32 + lrow) * E + khalf;       // A rows 0..15
  const float* xrow1 = xrow0 + (size_t)16 * E;                         // A rows 16..31
  const float* wrow0 = W + (size_t)(nt * 32 + lrow) * E + khalf;       // B cols 0..15
  const float* wrow1 = wrow0 + (size_t)16 * E;                         // B cols 16..31

  v8f c00 = {}, c01 = {}, c10 = {}, c11 = {};
#pragma unroll 2
  for (int kk = 0; kk < E; kk += 32) {
    const v16h a0 = cvt_row16(xrow0 + kk);
    const v16h a1 = cvt_row16(xrow1 + kk);
    const v16h b0 = cvt_row16(wrow0 + kk);
    const v16h b1 = cvt_row16(wrow1 + kk);
    c00 = wmma_f16(a0, b0, c00);
    c01 = wmma_f16(a0, b1, c01);
    c10 = wmma_f16(a1, b0, c10);
    c11 = wmma_f16(a1, b1, c11);
  }

  const float qscale = (z == 0) ? 0.125f : 1.0f;  // 1/sqrt(HD)

#pragma unroll
  for (int j = 0; j < 2; ++j) {
    const int col = nt * 32 + j * 16 + lrow;
    const float bv = bias[col];
    const int h = col >> 6;      // head
    const int d = col & 63;      // dim within head
#pragma unroll
    for (int i = 0; i < 2; ++i) {
      const v8f& c = (j == 0) ? (i == 0 ? c00 : c10) : (i == 0 ? c01 : c11);
#pragma unroll
      for (int r = 0; r < 8; ++r) {
        const int m  = mt * 32 + i * 16 + r + ((lane >> 4) << 3);
        const int s  = m >> 2;        // m / B
        const int b4 = m & 3;         // m % B
        const int bh = b4 * H + h;
        const _Float16 hv = (_Float16)((c[r] + bv) * qscale);
        if (z == 0)      Qw[((size_t)bh * S + s) * HD + d] = hv;
        else if (z == 1) Kw[((size_t)bh * S + s) * HD + d] = hv;
        else             Vt[((size_t)bh * HD + d) * S + s] = hv;
      }
    }
  }
}

// ---------------------------------------------------------------------------
// Kernel 2: flash attention, one wave per (bh, 16-query tile).
// Computes S^T = K * Q^T so each softmax row (query) lives in a fixed lane:
// row max/sum need only a single shfl_xor(16), and the P^T fragment for the
// PV WMMA is assembled with one xor-16 shuffle per element (no LDS).
// O^T accumulated as 4 independent d-tiles; scattered to Aout[(s*B+b)][e].
// ---------------------------------------------------------------------------
__global__ __launch_bounds__(128)
void flash_attn_kernel(const _Float16* __restrict__ Qw, const _Float16* __restrict__ Kw,
                       const _Float16* __restrict__ Vt, _Float16* __restrict__ Aout) {
  const int lane  = threadIdx.x & 31;
  const int wave  = threadIdx.x >> 5;
  const int qt    = blockIdx.x * 4 + wave;   // 0..127 query tile
  const int bh    = blockIdx.y;              // 0..31
  const int lrow  = lane & 15;
  const int khalf = (lane >> 4) * 16;
  const bool hiHalf = (lane >= 16);

  // Q^T as B operand: lane = query column, halves split the d (K) dimension.
  const _Float16* qp = Qw + ((size_t)bh * S + qt * 16 + lrow) * HD + khalf;
  const v16h bq0 = *(const v16h*)(qp);
  const v16h bq1 = *(const v16h*)(qp + 32);

  const _Float16* Krow = Kw + ((size_t)bh * S + lrow) * HD + khalf;  // + key*HD
  const _Float16* Vrow = Vt + ((size_t)bh * HD + lrow) * S + khalf;  // + dtile*16*S + kb

  v8f o0 = {}, o1 = {}, o2 = {}, o3 = {};
  float mrow = -INFINITY;   // running row max (per lane = per query, replicated)
  float lsum = 0.0f;        // running row sum

  for (int kb = 0; kb < S; kb += 32) {
    // ---- scores: two 16(key)x16(query) tiles of S^T, K-dim = HD = 64 ----
    const _Float16* kp0 = Krow + (size_t)kb * HD;
    const _Float16* kp1 = Krow + (size_t)(kb + 16) * HD;
    v8f s0 = {}, s1 = {};
    s0 = wmma_f16(*(const v16h*)(kp0),      bq0, s0);
    s1 = wmma_f16(*(const v16h*)(kp1),      bq0, s1);
    s0 = wmma_f16(*(const v16h*)(kp0 + 32), bq1, s0);
    s1 = wmma_f16(*(const v16h*)(kp1 + 32), bq1, s1);

    // ---- online softmax (query == lane&15 for every accumulator VGPR) ----
    float tmax = s0[0];
#pragma unroll
    for (int r = 1; r < 8; ++r) tmax = fmaxf(tmax, s0[r]);
#pragma unroll
    for (int r = 0; r < 8; ++r) tmax = fmaxf(tmax, s1[r]);
    tmax = fmaxf(tmax, __shfl_xor(tmax, 16, 32));
    const float mnew  = fmaxf(mrow, tmax);
    const float scale = __expf(mrow - mnew);
    mrow = mnew;

    float e0[8], e1[8];
    float rsum = 0.0f;
#pragma unroll
    for (int r = 0; r < 8; ++r) {
      e0[r] = __expf(s0[r] - mnew);
      e1[r] = __expf(s1[r] - mnew);
      rsum += e0[r] + e1[r];
    }
    rsum += __shfl_xor(rsum, 16, 32);
    lsum = lsum * scale + rsum;

#pragma unroll
    for (int r = 0; r < 8; ++r) {
      o0[r] *= scale; o1[r] *= scale; o2[r] *= scale; o3[r] *= scale;
    }

    // ---- build P^T B-fragment (32 keys x 16 queries) via xor-16 shuffles ----
    v16h pb;
#pragma unroll
    for (int r = 0; r < 8; ++r) {
      const float x0 = __shfl_xor(e0[r], 16, 32);
      const float x1 = __shfl_xor(e1[r], 16, 32);
      pb[r]     = (_Float16)(hiHalf ? x1    : e0[r]);  // keys kb+{0..7} / kb+{16..23}
      pb[8 + r] = (_Float16)(hiHalf ? e1[r] : x0);     // keys kb+{8..15} / kb+{24..31}
    }

    // ---- O^T += V^T(16d x 32k) * P^T(32k x 16q), 4 independent d-tiles ----
    const _Float16* vp = Vrow + kb;
    o0 = wmma_f16(*(const v16h*)(vp),          pb, o0);
    o1 = wmma_f16(*(const v16h*)(vp + 16 * S), pb, o1);
    o2 = wmma_f16(*(const v16h*)(vp + 32 * S), pb, o2);
    o3 = wmma_f16(*(const v16h*)(vp + 48 * S), pb, o3);
  }

  const float inv = 1.0f / lsum;
  const int s  = qt * 16 + lrow;   // query index == sequence position
  const int b4 = bh >> 3;
  const int h  = bh & 7;
  _Float16* orow = Aout + ((size_t)s * Bz + b4) * E + h * HD;
#pragma unroll
  for (int r = 0; r < 8; ++r) {
    const int db = r + (hiHalf ? 8 : 0);   // O^T C-layout: VGPR r holds d = db
    orow[ 0 + db] = (_Float16)(o0[r] * inv);
    orow[16 + db] = (_Float16)(o1[r] * inv);
    orow[32 + db] = (_Float16)(o2[r] * inv);
    orow[48 + db] = (_Float16)(o3[r] * inv);
  }
}

// ---------------------------------------------------------------------------
// Kernel 3: output projection (32x32 tile per wave), fp32 straight to d_out.
//   out[m][n] = Aout[m][:] . Wo[n][:] + bo[n],  m = s*B+b matches output layout.
// ---------------------------------------------------------------------------
__global__ __launch_bounds__(128)
void out_proj_kernel(const _Float16* __restrict__ Aout, const float* __restrict__ Wo,
                     const float* __restrict__ bo, float* __restrict__ out) {
  const int lane  = threadIdx.x & 31;
  const int wave  = threadIdx.x >> 5;
  const int mt    = blockIdx.x;               // 0..255
  const int nt    = blockIdx.y * 4 + wave;    // 0..15
  const int lrow  = lane & 15;
  const int khalf = (lane >> 4) * 16;

  const _Float16* arow0 = Aout + (size_t)(mt * 32 + lrow) * E + khalf;
  const _Float16* arow1 = arow0 + (size_t)16 * E;
  const float*    wrow0 = Wo + (size_t)(nt * 32 + lrow) * E + khalf;
  const float*    wrow1 = wrow0 + (size_t)16 * E;

  v8f c00 = {}, c01 = {}, c10 = {}, c11 = {};
#pragma unroll 2
  for (int kk = 0; kk < E; kk += 32) {
    const v16h a0 = *(const v16h*)(arow0 + kk);
    const v16h a1 = *(const v16h*)(arow1 + kk);
    const v16h b0 = cvt_row16(wrow0 + kk);
    const v16h b1 = cvt_row16(wrow1 + kk);
    c00 = wmma_f16(a0, b0, c00);
    c01 = wmma_f16(a0, b1, c01);
    c10 = wmma_f16(a1, b0, c10);
    c11 = wmma_f16(a1, b1, c11);
  }

#pragma unroll
  for (int j = 0; j < 2; ++j) {
    const int col = nt * 32 + j * 16 + lrow;
    const float bv = bo[col];
#pragma unroll
    for (int i = 0; i < 2; ++i) {
      const v8f& c = (j == 0) ? (i == 0 ? c00 : c10) : (i == 0 ? c01 : c11);
#pragma unroll
      for (int r = 0; r < 8; ++r) {
        const int m = mt * 32 + i * 16 + r + ((lane >> 4) << 3);
        out[(size_t)m * E + col] = c[r] + bv;
      }
    }
  }
}

// ---------------------------------------------------------------------------
extern "C" void kernel_launch(void* const* d_in, const int* in_sizes, int n_in,
                              void* d_out, int out_size, void* d_ws, size_t ws_size,
                              hipStream_t stream) {
  (void)in_sizes; (void)n_in; (void)out_size; (void)ws_size;
  const float* query = (const float*)d_in[0];
  const float* key   = (const float*)d_in[1];
  const float* value = (const float*)d_in[2];
  const float* ipw   = (const float*)d_in[3];
  const float* ipb   = (const float*)d_in[4];
  const float* opw   = (const float*)d_in[5];
  const float* opb   = (const float*)d_in[6];
  float* out = (float*)d_out;

  // Workspace layout (all f16): Q | K | V^T | attn-out  = 4 * 8 MB = 32 MB
  _Float16* Qw = (_Float16*)d_ws;
  _Float16* Kw = Qw + (size_t)NBH * S * HD;
  _Float16* Vt = Kw + (size_t)NBH * S * HD;
  _Float16* Ao = Vt + (size_t)NBH * S * HD;

  const dim3 blk(128);
  qkv_proj_kernel<<<dim3(M / 32, (E / 32) / 4, 3), blk, 0, stream>>>(
      query, key, value, ipw, ipb, Qw, Kw, Vt);
  flash_attn_kernel<<<dim3((S / 16) / 4, NBH), blk, 0, stream>>>(Qw, Kw, Vt, Ao);
  out_proj_kernel<<<dim3(M / 32, (E / 32) / 4), blk, 0, stream>>>(Ao, opw, opb, out);
}
